// PeepholeLSTM_38689065403008
// MI455X (gfx1250) — matrix operand order repaired and checked by reading, not compiled, not run on hardware
//
#include <hip/hip_runtime.h>
#include <hip/hip_bf16.h>

// Problem constants (PeepholeLSTM): B=64, S=1024, I=512, H=512
#define BATCH 64
#define SEQ   1024
#define IDIM  512
#define HDIM  512
#define GDIM  2048              // 4*H
#define MTOT  (BATCH * SEQ)     // 65536 rows of x

typedef __attribute__((ext_vector_type(16))) __bf16 v16bf;
typedef __attribute__((ext_vector_type(8)))  float  v8f;
typedef __attribute__((ext_vector_type(4)))  unsigned int u32x4;
typedef __attribute__((ext_vector_type(8)))  int i32x8;
typedef __attribute__((ext_vector_type(4)))  int i32x4;

union BF16x16 {
    v16bf v;
    uint4 q[2];
    __bf16 b[16];
};

// ---------------------------------------------------------------------------
// A-matrix fragment (16x32 bf16), source row-major [row][k], `stride` in elems.
// ISA layout: lanes 0-15 = rows 0-15 holding K {0..7,16..23};
//             lanes 16-31 = rows 0-15 holding K {8..15,24..31}.
// ---------------------------------------------------------------------------
__device__ __forceinline__ v16bf load_a_frag(const __bf16* base, int stride, int lane) {
    const int r  = lane & 15;
    const int hk = (lane >> 4) << 3;             // 0 or 8
    const __bf16* p = base + r * stride + hk;
    BF16x16 f;
    f.q[0] = *(const uint4*)(p);                 // K = hk .. hk+7
    f.q[1] = *(const uint4*)(p + 16);            // K = hk+16 .. hk+23
    return f.v;
}

// ---------------------------------------------------------------------------
// B-matrix fragment (32x16 bf16), source stored K-transposed row-major [n][k].
// ISA layout: lanes 0-15 = cols 0-15 holding K {0..7,16..23};
//             lanes 16-31 = cols 0-15 holding K {8..15,24..31}.
// ---------------------------------------------------------------------------
__device__ __forceinline__ v16bf load_b_frag(const __bf16* base, int stride, int lane) {
    const int n  = lane & 15;
    const int hk = (lane >> 4) << 3;             // 0 or 8
    const __bf16* p = base + n * stride + hk;
    BF16x16 f;
    f.q[0] = *(const uint4*)(p);                 // K = hk .. hk+7
    f.q[1] = *(const uint4*)(p + 16);            // K = hk+16 .. hk+23
    return f.v;
}

__device__ __forceinline__ float sigmoidf_(float x) {
    return 1.0f / (1.0f + __expf(-x));
}

// ---------------------------------------------------------------------------
// Tensor Data Mover: async 2D tile load (bf16) Global -> LDS.
// D# packing per CDNA5 ISA:
//   group0: [1:0]=count=1, [63:32]=lds_addr, [120:64]=global_addr, [127:126]=2
//   group1: [17:16]=data_size(1=2B), [79:48]=tensor_dim0, [111:80]=tensor_dim1,
//           [127:112]=tile_dim0, [143:128]=tile_dim1, [207:160]=dim0_stride
// Issued by one wave per workgroup; completion via s_wait_tensorcnt + barrier.
// ---------------------------------------------------------------------------
__device__ __forceinline__ void tdm_load_2d_bf16(
        unsigned int lds_off, const __bf16* gaddr,
        unsigned int tile_d0, unsigned int tile_d1,
        unsigned long long tensor_d0, unsigned long long tensor_d1,
        unsigned long long d0_stride) {
    unsigned long long ga = (unsigned long long)(uintptr_t)gaddr;
    u32x4 g0;
    g0[0] = 1u;                                          // count=1 (valid D#)
    g0[1] = lds_off;                                     // lds_addr (bytes)
    g0[2] = (unsigned int)(ga & 0xffffffffu);            // global_addr[31:0]
    g0[3] = (unsigned int)((ga >> 32) & 0x01ffffffu)     // global_addr[56:32]
          | (2u << 30);                                  // type = 2 ("image")
    i32x8 g1;
    g1[0] = (int)(1u << 16);                             // data_size = 1 -> 2B
    g1[1] = (int)((tensor_d0 & 0xffffu) << 16);          // tensor_dim0[15:0]
    g1[2] = (int)(((tensor_d0 >> 16) & 0xffffu)
          | ((tensor_d1 & 0xffffu) << 16));              // dim0 hi | dim1 lo
    g1[3] = (int)(((tensor_d1 >> 16) & 0xffffu)
          | ((tile_d0 & 0xffffu) << 16));                // dim1 hi | tile_dim0
    g1[4] = (int)(tile_d1 & 0xffffu);                    // tile_dim1 (dim2=0)
    g1[5] = (int)(d0_stride & 0xffffffffu);              // dim0_stride[31:0]
    g1[6] = (int)((d0_stride >> 32) & 0xffffu);          // dim0_stride[47:32]
    g1[7] = 0;                                           // dim1_stride unused (2D)
    i32x4 z4 = {};
#if defined(__clang_major__) && (__clang_major__ >= 23)
    i32x8 z8 = {};
    __builtin_amdgcn_tensor_load_to_lds(g0, g1, z4, z4, z8, 0);
#else
    __builtin_amdgcn_tensor_load_to_lds(g0, g1, z4, z4, 0);
#endif
}

// ---------------------------------------------------------------------------
// Phase 0a: x fp32 -> bf16 (vectorized), row-major [MTOT][512].
// ---------------------------------------------------------------------------
__global__ void convert_x_kernel(const float* __restrict__ x,
                                 __bf16* __restrict__ xbf) {
    size_t i4 = (size_t)blockIdx.x * blockDim.x + threadIdx.x;
    if (i4 >= (size_t)MTOT * IDIM / 4) return;
    float4 v = *(const float4*)(x + i4 * 4);
    union { uint2 u2; __bf16 b[4]; } p;
    p.b[0] = (__bf16)v.x; p.b[1] = (__bf16)v.y;
    p.b[2] = (__bf16)v.z; p.b[3] = (__bf16)v.w;
    *(uint2*)(xbf + i4 * 4) = p.u2;
}

// ---------------------------------------------------------------------------
// Phase 0b: W_xh fp32 (1024 x 2048) -> K-transposed bf16:
//   Wxt [n][k<512]  (phase-1 B tiles, TDM-friendly contiguous-K rows)
//   WtH [n][k<512]  (phase-2 B fragments, L2-resident)
// ---------------------------------------------------------------------------
__global__ void convert_w_kernel(const float* __restrict__ W_xh,
                                 __bf16* __restrict__ Wxt,
                                 __bf16* __restrict__ WtH) {
    int idx = blockIdx.x * blockDim.x + threadIdx.x;
    if (idx >= (IDIM + HDIM) * GDIM) return;
    int k = idx >> 11;            // row of W_xh
    int n = idx & (GDIM - 1);
    __bf16 b = (__bf16)W_xh[idx];
    if (k < IDIM) Wxt[(size_t)n * IDIM + k] = b;
    else          WtH[(size_t)n * HDIM + (k - IDIM)] = b;
}

// ---------------------------------------------------------------------------
// Phase 1: zx[65536 x 2048] = x_bf16 @ W_x_bf16 (fp32 accum).
// 256 threads (8 waves), tile 128x128. K chunks of 32 staged by the Tensor
// Data Mover into double-buffered LDS; wave 0 issues DMA + s_wait_tensorcnt,
// all waves only do fragment loads + WMMA.
// ---------------------------------------------------------------------------
__global__ void __launch_bounds__(256) zx_gemm_kernel(const __bf16* __restrict__ xbf,
                                                      const __bf16* __restrict__ Wxt,
                                                      float* __restrict__ zx) {
    __shared__ __bf16 Asub[2][128 * 32];    // [row m][k]  2 x 8 KB
    __shared__ __bf16 Btile[2][128 * 32];   // [row n][k]  2 x 8 KB

    const int Mbase = blockIdx.y * 128;
    const int Nbase = blockIdx.x * 128;
    const int tid  = threadIdx.x;
    const int lane = tid & 31;
    const int w    = tid >> 5;
    const int mrow0 = (w & 3) * 32;     // wave's row base inside tile
    const int ncol0 = (w >> 2) * 64;    // wave's col base inside tile

    v8f zero = {};
    v8f acc[2][4];
#pragma unroll
    for (int i = 0; i < 2; ++i)
#pragma unroll
        for (int j = 0; j < 4; ++j) acc[i][j] = zero;

    // prologue: DMA k-chunk 0 into buffer 0
    if (w == 0) {
        tdm_load_2d_bf16((unsigned int)(uintptr_t)&Asub[0][0],
                         xbf + (size_t)Mbase * IDIM, 32, 128,
                         IDIM, (unsigned long long)(MTOT - Mbase), IDIM);
        tdm_load_2d_bf16((unsigned int)(uintptr_t)&Btile[0][0],
                         Wxt + (size_t)Nbase * IDIM, 32, 128,
                         IDIM, (unsigned long long)(GDIM - Nbase), IDIM);
    }

    for (int step = 0; step < IDIM / 32; ++step) {
        const int buf = step & 1;
        if (w == 0) {
            if (step + 1 < IDIM / 32) {
                const int kk = (step + 1) * 32;
                tdm_load_2d_bf16((unsigned int)(uintptr_t)&Asub[buf ^ 1][0],
                                 xbf + (size_t)Mbase * IDIM + kk, 32, 128,
                                 IDIM, (unsigned long long)(MTOT - Mbase), IDIM);
                tdm_load_2d_bf16((unsigned int)(uintptr_t)&Btile[buf ^ 1][0],
                                 Wxt + (size_t)Nbase * IDIM + kk, 32, 128,
                                 IDIM, (unsigned long long)(GDIM - Nbase), IDIM);
                __builtin_amdgcn_s_wait_tensorcnt(2);  // current buffer landed
            } else {
                __builtin_amdgcn_s_wait_tensorcnt(0);  // last buffer landed
            }
        }
        __syncthreads();                               // LDS visible to all waves

        v16bf a0 = load_a_frag(&Asub[buf][(mrow0 + 0)  * 32], 32, lane);
        v16bf a1 = load_a_frag(&Asub[buf][(mrow0 + 16) * 32], 32, lane);
#pragma unroll
        for (int j = 0; j < 4; ++j) {
            v16bf b = load_b_frag(&Btile[buf][(ncol0 + j * 16) * 32], 32, lane);
            acc[0][j] = __builtin_amdgcn_wmma_f32_16x16x32_bf16(
                false, a0, false, b, (short)0, acc[0][j], false, false);
            acc[1][j] = __builtin_amdgcn_wmma_f32_16x16x32_bf16(
                false, a1, false, b, (short)0, acc[1][j], false, false);
        }
        __syncthreads();                               // buffer reusable for DMA
    }

    // ---- store C tiles: lane<16 -> M 0..7, lane>=16 -> M 8..15 ----
    const int n  = lane & 15;
    const int mb = (lane >> 4) * 8;
#pragma unroll
    for (int i = 0; i < 2; ++i) {
        int m0 = Mbase + mrow0 + i * 16;
#pragma unroll
        for (int j = 0; j < 4; ++j) {
            int col = Nbase + ncol0 + j * 16 + n;
#pragma unroll
            for (int r = 0; r < 8; ++r)
                zx[(size_t)(m0 + mb + r) * GDIM + col] = acc[i][j][r];
        }
    }
}

// ---------------------------------------------------------------------------
// Phase 2: recurrent scan. 4 persistent workgroups (16 batch rows each),
// 1024 threads = 32 waves. State h (bf16) + c (f32) + z (f32) + gate consts
// in 190 KB LDS (CDNA5: 320 KB/WGP). W_h streamed from L2 as B fragments.
// ---------------------------------------------------------------------------
__global__ void __launch_bounds__(1024) lstm_scan_kernel(
        const float* __restrict__ zx, const __bf16* __restrict__ WtH,
        const float* __restrict__ h0, const float* __restrict__ c0,
        const float* __restrict__ W_ci, const float* __restrict__ W_cf,
        const float* __restrict__ W_co, const float* __restrict__ b_i,
        const float* __restrict__ b_f, const float* __restrict__ b_c,
        const float* __restrict__ b_o, float* __restrict__ out) {
    extern __shared__ char smem[];
    float*  zbuf = (float*)smem;                  // 16 x 2048 f32 = 128 KB
    float*  cst  = zbuf + 16 * GDIM;              // 16 x 512  f32 =  32 KB
    __bf16* hbf  = (__bf16*)(cst + 16 * HDIM);    // 16 x 512 bf16 =  16 KB
    float*  par  = (float*)(hbf + 16 * HDIM);     // 7 x 512  f32  =  14 KB

    const int bBase = blockIdx.x * 16;
    const int tid  = threadIdx.x;
    const int lane = tid & 31;
    const int w    = tid >> 5;                    // 0..31
    const int n0   = w * 64;                      // wave's 64 output columns

    float* hseq = out;                                   // B*S*H
    float* hT   = out + (size_t)BATCH * SEQ * HDIM;      // B*H
    float* cT   = hT + (size_t)BATCH * HDIM;             // B*H

    // gate constants -> LDS (read every step by every thread)
    for (int j = tid; j < HDIM; j += 1024) {
        par[0 * HDIM + j] = W_ci[j];
        par[1 * HDIM + j] = W_cf[j];
        par[2 * HDIM + j] = W_co[j];
        par[3 * HDIM + j] = b_i[j];
        par[4 * HDIM + j] = b_f[j];
        par[5 * HDIM + j] = b_c[j];
        par[6 * HDIM + j] = b_o[j];
    }
    // init state from h0/c0
    for (int u = tid; u < 16 * HDIM; u += 1024) {
        int m = u >> 9, j = u & (HDIM - 1);
        cst[u] = c0[(size_t)(bBase + m) * HDIM + j];
        hbf[u] = (__bf16)h0[(size_t)(bBase + m) * HDIM + j];
    }
    __syncthreads();

    v8f zero = {};
    for (int t = 0; t < SEQ; ++t) {
        // ---- z = h @ W_h over this wave's 4 column tiles ----
        v8f acc[4];
#pragma unroll
        for (int j = 0; j < 4; ++j) acc[j] = zero;

        for (int kk = 0; kk < HDIM; kk += 32) {
            v16bf a = load_a_frag(hbf + kk, HDIM, lane);
            __builtin_prefetch(WtH + (size_t)n0 * HDIM + ((kk + 32) & (HDIM - 1)), 0, 1);
#pragma unroll
            for (int j = 0; j < 4; ++j) {
                const __bf16* bb = WtH + (size_t)(n0 + j * 16) * HDIM + kk;
                v16bf b = load_b_frag(bb, HDIM, lane);
                acc[j] = __builtin_amdgcn_wmma_f32_16x16x32_bf16(
                    false, a, false, b, (short)0, acc[j], false, false);
            }
        }

        // ---- z += zx[:, t, :]; spill to LDS ----
        {
            const int n  = lane & 15;
            const int mb = (lane >> 4) * 8;
#pragma unroll
            for (int j = 0; j < 4; ++j) {
                int col = n0 + j * 16 + n;
#pragma unroll
                for (int r = 0; r < 8; ++r) {
                    int m = mb + r;
                    float zv = acc[j][r] +
                        zx[((size_t)(bBase + m) * SEQ + t) * GDIM + col];
                    zbuf[m * GDIM + col] = zv;
                }
            }
        }
        __syncthreads();

        // ---- gates (TRANS ops: v_exp/v_tanh co-execute with matrix pipe) ----
        for (int u = tid; u < 16 * HDIM; u += 1024) {
            int m = u >> 9, j = u & (HDIM - 1);
            float zi = zbuf[m * GDIM + j];
            float zf = zbuf[m * GDIM + HDIM + j];
            float zc = zbuf[m * GDIM + 2 * HDIM + j];
            float zo = zbuf[m * GDIM + 3 * HDIM + j];
            float c_old = cst[u];
            float it = sigmoidf_(zi + c_old * par[0 * HDIM + j] + par[3 * HDIM + j]);
            float ft = sigmoidf_(zf + c_old * par[1 * HDIM + j] + par[4 * HDIM + j]);
            float cn = ft * c_old + it * tanhf(zc + par[5 * HDIM + j]);
            float ot = sigmoidf_(zo + cn * par[2 * HDIM + j] + par[6 * HDIM + j]);
            float hn = ot * tanhf(cn);
            cst[u] = cn;
            hbf[u] = (__bf16)hn;
            hseq[((size_t)(bBase + m) * SEQ + t) * HDIM + j] = hn;
            if (t == SEQ - 1) {
                hT[(size_t)(bBase + m) * HDIM + j] = hn;
                cT[(size_t)(bBase + m) * HDIM + j] = cn;
            }
        }
        __syncthreads();
    }
}

// ---------------------------------------------------------------------------
extern "C" void kernel_launch(void* const* d_in, const int* in_sizes, int n_in,
                              void* d_out, int out_size, void* d_ws, size_t ws_size,
                              hipStream_t stream) {
    const float* x    = (const float*)d_in[0];
    const float* h0   = (const float*)d_in[1];
    const float* c0   = (const float*)d_in[2];
    const float* W_xh = (const float*)d_in[3];
    const float* W_ci = (const float*)d_in[4];
    const float* W_cf = (const float*)d_in[5];
    const float* W_co = (const float*)d_in[6];
    const float* b_i  = (const float*)d_in[7];
    const float* b_f  = (const float*)d_in[8];
    const float* b_c  = (const float*)d_in[9];
    const float* b_o  = (const float*)d_in[10];
    float* out = (float*)d_out;

    // workspace layout
    char* ws = (char*)d_ws;
    float*  zx  = (float*)ws;                                           // 512 MiB
    __bf16* xbf = (__bf16*)(ws + (size_t)MTOT * GDIM * sizeof(float));  //  64 MiB
    __bf16* Wxt = (__bf16*)((char*)xbf + (size_t)MTOT * IDIM * sizeof(__bf16));
    __bf16* WtH = (__bf16*)((char*)Wxt + (size_t)GDIM * IDIM * sizeof(__bf16));

    // phase 0: x -> bf16; W_x / W_h -> K-transposed bf16
    {
        size_t n4 = (size_t)MTOT * IDIM / 4;
        convert_x_kernel<<<(unsigned)((n4 + 255) / 256), 256, 0, stream>>>(x, xbf);
        int nw = (IDIM + HDIM) * GDIM;
        convert_w_kernel<<<(nw + 255) / 256, 256, 0, stream>>>(W_xh, Wxt, WtH);
    }
    // phase 1: zx = x @ W_x (TDM-staged bf16 WMMA, fp32 accum)
    {
        dim3 grid(GDIM / 128, MTOT / 128);   // 16 x 512 blocks
        zx_gemm_kernel<<<grid, 256, 0, stream>>>(xbf, Wxt, zx);
    }
    // phase 2: recurrent scan, 4 persistent workgroups, 190 KB dynamic LDS
    {
        size_t shmem = (size_t)16 * GDIM * sizeof(float)     // zbuf
                     + (size_t)16 * HDIM * sizeof(float)     // c state
                     + (size_t)16 * HDIM * sizeof(__bf16)    // h state (bf16)
                     + (size_t)7 * HDIM * sizeof(float);     // gate constants
        lstm_scan_kernel<<<4, 1024, shmem, stream>>>(
            zx, WtH, h0, c0, W_ci, W_cf, W_co, b_i, b_f, b_c, b_o, out);
    }
}
